// ASPModel_85298050498763
// MI455X (gfx1250) — matrix operand order repaired and checked by reading, compiled, not run on hardware
//
#include <hip/hip_runtime.h>

typedef __attribute__((ext_vector_type(16))) _Float16 v16h;
typedef __attribute__((ext_vector_type(8)))  float    v8f;

#define T_SEQ 288
#define B_DIM 2048
#define D_DIM 200
#define N_EMB 300
#define MLP_H 2048

// ---------------- small prep kernels ----------------

// proj[e][g] = dot(emb[e,:], W_ih0[g,:])  -- folds the [B*T,200]x[200,4] GEMM
// into a 300x4 lookup table (embedding table has only 300 rows).
__global__ void proj_kernel(const float* __restrict__ emb,
                            const float* __restrict__ Wih0,
                            float* __restrict__ proj) {
  int i = blockIdx.x * blockDim.x + threadIdx.x;
  if (i >= N_EMB * 4) return;
  int e = i >> 2, g = i & 3;
  const float* ev = emb + e * D_DIM;
  const float* wv = Wih0 + g * D_DIM;
  float s = 0.f;
  for (int d = 0; d < D_DIM; ++d) s += ev[d] * wv[d];
  proj[i] = s;
}

__global__ void f32_to_f16_kernel(const float* __restrict__ s,
                                  _Float16* __restrict__ d, int n) {
  int i = blockIdx.x * blockDim.x + threadIdx.x;
  if (i < n) d[i] = (_Float16)s[i];
}

__device__ __forceinline__ float sigf(float x) { return 1.f / (1.f + __expf(-x)); }

// ---------------- fused 2-layer LSTM (H=1) + ReLU ----------------
// One lane per batch element; sequential scan over T. Writes relu(h2) as f16
// (the A operand of GEMM1).
__global__ __launch_bounds__(256) void lstm_kernel(
    const int* __restrict__ x, const float* __restrict__ proj,
    const float* __restrict__ Whh0,
    const float* __restrict__ bih0, const float* __restrict__ bhh0,
    const float* __restrict__ Wih1, const float* __restrict__ Whh1,
    const float* __restrict__ bih1, const float* __restrict__ bhh1,
    _Float16* __restrict__ A1) {
  int b = blockIdx.x * blockDim.x + threadIdx.x;
  if (b >= B_DIM) return;
  float w0[4], wi1[4], w1[4], bb0[4], bb1[4];
  for (int g = 0; g < 4; ++g) {
    w0[g] = Whh0[g];
    wi1[g] = Wih1[g];
    w1[g] = Whh1[g];
    bb0[g] = bih0[g] + bhh0[g];
    bb1[g] = bih1[g] + bhh1[g];
  }
  float h0 = 0.f, c0 = 0.f, h1 = 0.f, c1 = 0.f;
  const int* xb = x + (size_t)b * T_SEQ;
  _Float16* ob = A1 + (size_t)b * T_SEQ;
  for (int t = 0; t < T_SEQ; ++t) {
    int e = xb[t];
    const float* p = proj + e * 4;
    // layer 0, PyTorch gate order i,f,g,o
    float gi = p[0] + bb0[0] + h0 * w0[0];
    float gf = p[1] + bb0[1] + h0 * w0[1];
    float gg = p[2] + bb0[2] + h0 * w0[2];
    float go = p[3] + bb0[3] + h0 * w0[3];
    c0 = sigf(gf) * c0 + sigf(gi) * tanhf(gg);
    h0 = sigf(go) * tanhf(c0);
    // layer 1 (input dim 1)
    float ai = h0 * wi1[0] + bb1[0] + h1 * w1[0];
    float af = h0 * wi1[1] + bb1[1] + h1 * w1[1];
    float ag = h0 * wi1[2] + bb1[2] + h1 * w1[2];
    float ao = h0 * wi1[3] + bb1[3] + h1 * w1[3];
    c1 = sigf(af) * c1 + sigf(ai) * tanhf(ag);
    h1 = sigf(ao) * tanhf(c1);
    float r = h1 > 0.f ? h1 : 0.f;  // fused relu
    ob[t] = (_Float16)r;
  }
}

// ---------------- WMMA GEMM: D = epi(A[M,K] * B[N,K]^T + bias) ----------------
// 256 threads = 8 waves, wave grid WM x WN (WM*WN == 8). Each wave computes a
// 32x32 output tile: 4 v_wmma_f32_16x16x32_f16 per K-step from 8 ds_load_b128
// (A frags reused across 2 B tiles and vice versa -> 2 LDS reads per WMMA).
// Block tile (32*WM) x (32*WN), BK=32, double-buffered LDS with 40-half row
// stride for bank spread. EPI==0: relu -> f16 store. EPI==1: sigmoid -> f32.

union FragU { v16h v; uint4 q[2]; };

template <int EPI, int WM, int WN>
__global__ __launch_bounds__(256) void gemm_wmma_kernel(
    const _Float16* __restrict__ A, const _Float16* __restrict__ B,
    const float* __restrict__ bias, void* __restrict__ Dst, int N, int K) {
  constexpr int BM = 32 * WM, BN = 32 * WN, BK = 32, LD = 40;
  constexpr int NACH = (BM * 4) / 256;  // A b128-chunks per thread
  __shared__ __align__(16) _Float16 sA[2][BM * LD];
  __shared__ __align__(16) _Float16 sB[2][BN * LD];

  const int tid = threadIdx.x;
  const int lane = tid & 31;
  const int wave = tid >> 5;
  const int wm = wave / WN;
  const int wn = wave % WN;
  const int bm = blockIdx.x, bn = blockIdx.y;

  // staging assignment: chunk c -> row c>>2, k-offset (c&3)*8 (halves)
  const _Float16* pA[NACH];
  int ldsA[NACH];
#pragma unroll
  for (int i = 0; i < NACH; ++i) {
    const int c = tid + i * 256;
    const int r = c >> 2, kp = (c & 3) * 8;
    pA[i] = A + (size_t)(bm * BM + r) * K + kp;
    ldsA[i] = r * LD + kp;
  }
  const int rB = tid >> 2, kpB = (tid & 3) * 8;
  const _Float16* pB = B + (size_t)(bn * BN + rB) * K + kpB;
  const int ldsB = rB * LD + kpB;
  const bool doB = (tid < BN * 4);

  const int nk = K / BK;

  uint4 ar[NACH];
  uint4 br = make_uint4(0u, 0u, 0u, 0u);
#pragma unroll
  for (int i = 0; i < NACH; ++i) ar[i] = *(const uint4*)(pA[i]);
  if (doB) br = *(const uint4*)(pB);
#pragma unroll
  for (int i = 0; i < NACH; ++i) *(uint4*)&sA[0][ldsA[i]] = ar[i];
  if (doB) *(uint4*)&sB[0][ldsB] = br;
  __syncthreads();

  v8f acc[4] = {};

  // fragment addressing (documented gfx1250 16-bit WMMA layouts, wave32)
  const int mrow = wm * 32 + (lane & 15);    // A: row = lane%16 within tile
  const int abase = (lane >> 4) << 3;        // A: K-group base 0 or 8
  const int ncol = wn * 32 + (lane & 15);    // B: col = lane%16
  const int kb0 = (lane >> 4) << 3;          // B: K-group base 0 or 8

  for (int kk = 0; kk < nk; ++kk) {
    const int cur = kk & 1;
    if (kk + 1 < nk) {  // fetch next K tile to registers while computing
      const int ko = (kk + 1) * BK;
#pragma unroll
      for (int i = 0; i < NACH; ++i) ar[i] = *(const uint4*)(pA[i] + ko);
      if (doB) br = *(const uint4*)(pB + ko);
    }
    if (kk + 2 < nk) {  // global_prefetch_b8 the tile after that
      __builtin_prefetch(pA[0] + (size_t)(kk + 2) * BK, 0, 1);
      if (doB) __builtin_prefetch(pB + (size_t)(kk + 2) * BK, 0, 1);
    }

    FragU fa0, fa1, fb0, fb1;
    fa0.q[0] = *(const uint4*)&sA[cur][mrow * LD + abase];
    fa0.q[1] = *(const uint4*)&sA[cur][mrow * LD + 16 + abase];
    fa1.q[0] = *(const uint4*)&sA[cur][(mrow + 16) * LD + abase];
    fa1.q[1] = *(const uint4*)&sA[cur][(mrow + 16) * LD + 16 + abase];
    fb0.q[0] = *(const uint4*)&sB[cur][ncol * LD + kb0];
    fb0.q[1] = *(const uint4*)&sB[cur][ncol * LD + kb0 + 16];
    fb1.q[0] = *(const uint4*)&sB[cur][(ncol + 16) * LD + kb0];
    fb1.q[1] = *(const uint4*)&sB[cur][(ncol + 16) * LD + kb0 + 16];

    acc[0] = __builtin_amdgcn_wmma_f32_16x16x32_f16(
        false, fa0.v, false, fb0.v, (short)0, acc[0], false, false);
    acc[1] = __builtin_amdgcn_wmma_f32_16x16x32_f16(
        false, fa0.v, false, fb1.v, (short)0, acc[1], false, false);
    acc[2] = __builtin_amdgcn_wmma_f32_16x16x32_f16(
        false, fa1.v, false, fb0.v, (short)0, acc[2], false, false);
    acc[3] = __builtin_amdgcn_wmma_f32_16x16x32_f16(
        false, fa1.v, false, fb1.v, (short)0, acc[3], false, false);

    if (kk + 1 < nk) {
      const int nxt = cur ^ 1;
#pragma unroll
      for (int i = 0; i < NACH; ++i) *(uint4*)&sA[nxt][ldsA[i]] = ar[i];
      if (doB) *(uint4*)&sB[nxt][ldsB] = br;
    }
    __syncthreads();
  }

  // epilogue: C/D layout -> lane L, vgpr j holds (m = j + (L>=16)*8, n = L%16)
  const int rsub = (lane >> 4) << 3;
#pragma unroll
  for (int i = 0; i < 2; ++i) {
#pragma unroll
    for (int jn = 0; jn < 2; ++jn) {
      const int rb = bm * BM + wm * 32 + i * 16 + rsub;
      const int col = bn * BN + wn * 32 + jn * 16 + (lane & 15);
      const float bv = bias[col];
      const v8f a = acc[i * 2 + jn];
#pragma unroll
      for (int j = 0; j < 8; ++j) {
        float v = a[j] + bv;
        if (EPI == 0) {
          v = v > 0.f ? v : 0.f;
          ((_Float16*)Dst)[(size_t)(rb + j) * N + col] = (_Float16)v;
        } else {
          ((float*)Dst)[(size_t)(rb + j) * N + col] = 1.f / (1.f + __expf(-v));
        }
      }
    }
  }
}

// ---------------- launcher ----------------

extern "C" void kernel_launch(void* const* d_in, const int* in_sizes, int n_in,
                              void* d_out, int out_size, void* d_ws, size_t ws_size,
                              hipStream_t stream) {
  const int* x = (const int*)d_in[0];
  const float* emb = (const float*)d_in[1];
  const float* W_ih0 = (const float*)d_in[2];
  const float* W_hh0 = (const float*)d_in[3];
  const float* b_ih0 = (const float*)d_in[4];
  const float* b_hh0 = (const float*)d_in[5];
  const float* W_ih1 = (const float*)d_in[6];
  const float* W_hh1 = (const float*)d_in[7];
  const float* b_ih1 = (const float*)d_in[8];
  const float* b_hh1 = (const float*)d_in[9];
  const float* W1 = (const float*)d_in[10];
  const float* b1 = (const float*)d_in[11];
  const float* W2 = (const float*)d_in[12];
  const float* b2 = (const float*)d_in[13];
  float* out = (float*)d_out;

  // workspace carve (256B-aligned chunks)
  char* w = (char*)d_ws;
  float* proj = (float*)w;          w += 4864;                       // 300*4 f32
  _Float16* A1 = (_Float16*)w;      w += (size_t)B_DIM * T_SEQ * 2;  // relu(h2) f16
  _Float16* W1h = (_Float16*)w;     w += (size_t)MLP_H * T_SEQ * 2;
  _Float16* W2h = (_Float16*)w;     w += (size_t)T_SEQ * MLP_H * 2;
  _Float16* hidden = (_Float16*)w;  // [B_DIM, MLP_H] f16

  const int nW = MLP_H * T_SEQ;  // 589824

  proj_kernel<<<(N_EMB * 4 + 255) / 256, 256, 0, stream>>>(emb, W_ih0, proj);
  f32_to_f16_kernel<<<(nW + 255) / 256, 256, 0, stream>>>(W1, W1h, nW);
  f32_to_f16_kernel<<<(nW + 255) / 256, 256, 0, stream>>>(W2, W2h, nW);
  lstm_kernel<<<B_DIM / 256, 256, 0, stream>>>(x, proj, W_hh0, b_ih0, b_hh0,
                                               W_ih1, W_hh1, b_ih1, b_hh1, A1);
  // hidden = relu(A1 @ W1^T + b1)   M=2048, N=2048, K=288
  // waves 4x2 -> block tile 128x64, grid 16x32
  gemm_wmma_kernel<0, 4, 2><<<dim3(B_DIM / 128, MLP_H / 64), 256, 0, stream>>>(
      A1, W1h, b1, hidden, MLP_H, T_SEQ);
  // out = sigmoid(hidden @ W2^T + b2)   M=2048, N=288, K=2048
  // waves 8x1 -> block tile 256x32, grid 8x9 (288 = 9*32)
  gemm_wmma_kernel<1, 8, 1><<<dim3(B_DIM / 256, T_SEQ / 32), 256, 0, stream>>>(
      hidden, W2h, b2, out, T_SEQ, MLP_H);
}
